// KNNLayer_20255065768644
// MI455X (gfx1250) — compile-verified, matrix operand order checked by reading, not run on hardware
//
#include <hip/hip_runtime.h>
#include <stdint.h>

typedef __attribute__((ext_vector_type(8))) int v8i;

#define NPTS 65536
#define KNN  16
#define NCH  64

// Pack int32 coords [N,3] (values 0..127) into one u32 per point: x | y<<8 | z<<16.
__global__ void pack_coords_kernel(const int* __restrict__ coords,
                                   uint32_t* __restrict__ dbpack) {
    int i = blockIdx.x * blockDim.x + threadIdx.x;
    if (i < NPTS) {
        uint32_t x = (uint32_t)coords[3 * i + 0] & 0xFFu;
        uint32_t y = (uint32_t)coords[3 * i + 1] & 0xFFu;
        uint32_t z = (uint32_t)coords[3 * i + 2] & 0xFFu;
        dbpack[i] = x | (y << 8) | (z << 16);
    }
}

__device__ __forceinline__ void insert_key(uint32_t key, uint32_t lst[16]) {
    if (key < lst[15]) {
        uint32_t cur = key;
#pragma unroll
        for (int i = 0; i < 16; ++i) {
            uint32_t lo = (cur < lst[i]) ? cur : lst[i];
            uint32_t hh = (cur < lst[i]) ? lst[i] : cur;
            lst[i] = lo;
            cur    = hh;
        }
    }
}

// One wave (32 threads) per 16-query tile. Exact integer KNN via iu8 WMMA dot
// products. Software-pipelined: selection for tile-pair (tp-1) runs in the
// co-exec shadow of tile-pair tp's two WMMAs (double-buffered LDS keys), and
// the dbpack global loads run one iteration ahead.
__global__ __launch_bounds__(32) void knn_gather_kernel(
    const int* __restrict__ coords,
    const float* __restrict__ features,
    const uint32_t* __restrict__ dbpack,
    float* __restrict__ out) {

    __shared__ int      s_qsq[16];
    __shared__ uint32_t s_tile[2 * 512];   // double-buffered keys, [buf][query][col 0..31]
    __shared__ uint32_t s_merge[16 * 16];  // odd-lane lists for final merge
    __shared__ int      s_idx[16 * 16];    // final neighbor indices

    const int lane  = (int)threadIdx.x;    // 0..31
    const int qbase = (int)blockIdx.x * 16;
    const int half  = lane >> 4;           // 0: rows 0-7, 1: rows 8-15 (C/D layout)
    const int col   = lane & 15;           // D-matrix column this lane holds

    // ---- load + pack the 16 query points, compute |q|^2 ----
    uint32_t qpack = 0;
    if (lane < 16) {
        int x = coords[3 * (qbase + lane) + 0];
        int y = coords[3 * (qbase + lane) + 1];
        int z = coords[3 * (qbase + lane) + 2];
        qpack = (uint32_t)(x | (y << 8) | (z << 16));
        s_qsq[lane] = x * x + y * y + z * z;
    }
    __syncthreads();

    int qsq8[8];
#pragma unroll
    for (int v = 0; v < 8; ++v) qsq8[v] = s_qsq[v + 8 * half];

    // A-matrix (16x64 u8): lanes 0-15, VGPR0 holds K=0..3 bytes = (x,y,z,0).
    v8i amat = {};
    amat[0] = (lane < 16) ? (int)qpack : 0;

    // Sorted ascending top-16 list of packed keys (dist<<16 | idx).
    uint32_t lst[16];
#pragma unroll
    for (int i = 0; i < 16; ++i) lst[i] = 0xFFFFFFFFu;

    const int mq  = lane >> 1;         // query owned by this lane pair
    const int cb2 = (lane & 1) * 16;   // even lane: tile0 cols, odd lane: tile1 cols
    const int NT  = NPTS / 32;

    // Pipelined db loads: (p0, p1) always holds tile-pair tp's data.
    uint32_t p0 = dbpack[col];
    uint32_t p1 = dbpack[16 + col];

    for (int tp = 0; tp < NT; ++tp) {
        const int base = tp * 32;

        // ---- issue next iteration's loads (clamped at the end, no divergence) ----
        const int nbase = (tp + 1 < NT) ? base + 32 : base;
        uint32_t np0 = dbpack[nbase + col];
        uint32_t np1 = dbpack[nbase + 16 + col];
        __builtin_prefetch(dbpack + base + 1024, 0, 0);  // global_prefetch_b8

        // ---- issue both WMMAs for tile-pair tp ----
        v8i b0 = {};
        v8i b1 = {};
        b0[0] = (lane < 16) ? (int)p0 : 0;
        b1[0] = (lane < 16) ? (int)p1 : 0;
        v8i d0 = {};
        v8i d1 = {};
        d0 = __builtin_amdgcn_wmma_i32_16x16x64_iu8(false, amat, false, b0, d0, false, false);
        d1 = __builtin_amdgcn_wmma_i32_16x16x64_iu8(false, amat, false, b1, d1, false, false);

        // ---- independent work fills the IU8 WMMA->VALU shadow ----
        int c0x = (int)(p0 & 0xFFu), c0y = (int)((p0 >> 8) & 0xFFu), c0z = (int)((p0 >> 16) & 0xFFu);
        int c1x = (int)(p1 & 0xFFu), c1y = (int)((p1 >> 8) & 0xFFu), c1z = (int)((p1 >> 16) & 0xFFu);
        int csq0 = c0x * c0x + c0y * c0y + c0z * c0z;
        int csq1 = c1x * c1x + c1y * c1y + c1z * c1z;

        // Selection over the keys staged by the PREVIOUS iteration (independent
        // of d0/d1): 16 ds_loads + guarded insertion chains per lane.
        if (tp > 0) {
            const uint32_t* prev = s_tile + (((tp - 1) & 1) << 9);
#pragma unroll
            for (int j = 0; j < 16; ++j) {
                insert_key(prev[mq * 32 + cb2 + j], lst);
            }
        }

        // ---- now consume d0/d1: pack keys, stage into the other buffer ----
        // dist(m, n) = |q_m|^2 + |c_n|^2 - 2 * <q_m, c_n>   (exact, < 2^16)
        uint32_t* curb = s_tile + ((tp & 1) << 9);
        uint32_t idx0 = (uint32_t)(base + col);
        uint32_t idx1 = (uint32_t)(base + 16 + col);
#pragma unroll
        for (int v = 0; v < 8; ++v) {
            int m = v + 8 * half;
            int dist0 = qsq8[v] + csq0 - 2 * d0[v];
            int dist1 = qsq8[v] + csq1 - 2 * d1[v];
            curb[m * 32 + col]      = ((uint32_t)dist0 << 16) | idx0;
            curb[m * 32 + 16 + col] = ((uint32_t)dist1 << 16) | idx1;
        }
        __syncthreads();

        p0 = np0;
        p1 = np1;
    }

    // ---- epilogue: selection over the final staged buffer ----
    {
        const uint32_t* prev = s_tile + (((NT - 1) & 1) << 9);
#pragma unroll
        for (int j = 0; j < 16; ++j) {
            insert_key(prev[mq * 32 + cb2 + j], lst);
        }
    }

    // ---- merge the two half-streams per query ----
    if (lane & 1) {
#pragma unroll
        for (int i = 0; i < 16; ++i) s_merge[mq * 16 + i] = lst[i];
    }
    __syncthreads();
    if (!(lane & 1)) {
#pragma unroll
        for (int j = 0; j < 16; ++j) {
            insert_key(s_merge[mq * 16 + j], lst);
        }
#pragma unroll
        for (int i = 0; i < 16; ++i) s_idx[mq * 16 + i] = (int)(lst[i] & 0xFFFFu);
    }
    __syncthreads();

    // ---- gather + sum neighbor features: 2 channels per lane ----
    const float2* __restrict__ f2 = (const float2*)features;
    for (int q = 0; q < 16; ++q) {
        float2 acc;
        acc.x = 0.0f;
        acc.y = 0.0f;
#pragma unroll
        for (int j = 0; j < KNN; ++j) {
            int ni = s_idx[q * 16 + j];
            float2 v = f2[(size_t)ni * (NCH / 2) + lane];
            acc.x += v.x;
            acc.y += v.y;
        }
        float2* o2 = (float2*)(out + (size_t)(qbase + q) * NCH);
        o2[lane] = acc;
    }
}

extern "C" void kernel_launch(void* const* d_in, const int* in_sizes, int n_in,
                              void* d_out, int out_size, void* d_ws, size_t ws_size,
                              hipStream_t stream) {
    const int*   coords   = (const int*)d_in[0];    // [N,3] int32
    const float* features = (const float*)d_in[1];  // [N,64] f32
    float*       out      = (float*)d_out;          // [N,64] f32
    uint32_t*    dbpack   = (uint32_t*)d_ws;        // N u32 (256 KB scratch)

    pack_coords_kernel<<<NPTS / 256, 256, 0, stream>>>(coords, dbpack);
    knn_gather_kernel<<<NPTS / 16, 32, 0, stream>>>(coords, features, dbpack, out);
}
